// MultiQueryAttention_41798621725182
// MI455X (gfx1250) — matrix operand order repaired and checked by reading, not compiled, not run on hardware
//
#include <hip/hip_runtime.h>

// ---------------------------------------------------------------------------
// MQA on MI455X (gfx1250, wave32, WMMA).
// B=2, S=2048, D=1024, H=16, DH=64. ~53 GFLOP vs ~75MB HBM traffic =>
// matrix-core bound: every matmul goes through v_wmma_f32_16x16x32_bf16.
// softmax scale 1/sqrt(64)=2^-3 is folded (exactly) into the Q projection.
// Workspace: ~46 MiB bf16 intermediates; fully L2-resident (192 MB).
// ---------------------------------------------------------------------------

#define BB  2
#define SS  2048
#define DD  1024
#define NH  16
#define DH  64
#define MM  (BB * SS)      // 4096 flattened rows

typedef __attribute__((ext_vector_type(16))) __bf16        v16bf;
typedef __attribute__((ext_vector_type(8)))  float         v8f;
typedef __attribute__((ext_vector_type(4)))  unsigned int  v4u;

union Frag {
    v16bf v;
    v4u   q[2];
};

__device__ __forceinline__ unsigned short f2bf(float f) {
    // round-to-nearest-even fp32 -> bf16
    unsigned int u = __float_as_uint(f);
    unsigned int r = u + 0x7FFFu + ((u >> 16) & 1u);
    return (unsigned short)(r >> 16);
}

// A-matrix fragment, 16x32 tile of a row-major bf16 matrix (ld in elements).
// ISA layout (16-bit A 16x32): lane<16 holds row M=lane, K={0..7,16..23};
// lane>=16 holds same row, K={8..15,24..31}. Two 16B loads per lane.
__device__ __forceinline__ v16bf load_a_frag(const unsigned short* base, int ld,
                                             int row0, int k0, int lane) {
    int row  = row0 + (lane & 15);
    int half = lane >> 4;
    const unsigned short* p = base + (size_t)row * ld + k0;
    Frag f;
    f.q[0] = *(const v4u*)(p + half * 8);
    f.q[1] = *(const v4u*)(p + 16 + half * 8);
    return f.v;
}

// B-matrix fragment, 32x16 tile where B[k][n] = Wrow[n][k], Wrow row-major
// (n, k). ISA layout: lane<16 holds column N=lane, K={0..7,16..23}; lane>=16 holds
// N=lane-16, K={8..15,24..31}. Two 16B loads per lane.
__device__ __forceinline__ v16bf load_b_frag(const unsigned short* base, int ld,
                                             int n0, int k0, int lane) {
    int n    = n0 + (lane & 15);
    int half = lane >> 4;
    const unsigned short* p = base + (size_t)n * ld + k0 + half * 8;
    Frag f;
    f.q[0] = *(const v4u*)(p);
    f.q[1] = *(const v4u*)(p + 16);
    return f.v;
}

__device__ __forceinline__ v8f wmma_bf16(v16bf a, v16bf b, v8f c) {
    return __builtin_amdgcn_wmma_f32_16x16x32_bf16(false, a, false, b,
                                                   (short)0, c, false, false);
}

// ---------------------------------------------------------------------------
// vectorized fp32 -> bf16 (4 elements/thread; all sizes are multiples of 4)
// ---------------------------------------------------------------------------
__global__ __launch_bounds__(256) void k_f32_to_bf16x4(const float4* __restrict__ in,
                                                       uint2* __restrict__ out,
                                                       int n4) {
    int i = blockIdx.x * 256 + threadIdx.x;
    if (i < n4) {
        float4 f = in[i];
        uint2 o;
        o.x = (unsigned)f2bf(f.x) | ((unsigned)f2bf(f.y) << 16);
        o.y = (unsigned)f2bf(f.z) | ((unsigned)f2bf(f.w) << 16);
        out[i] = o;
    }
}

// ---------------------------------------------------------------------------
// bf16 transpose: vp [B,S,DH] -> vpT [B,DH,S]  (makes PV B-frags contiguous)
// ---------------------------------------------------------------------------
__global__ __launch_bounds__(256) void k_transpose_vp(const unsigned short* __restrict__ vp,
                                                      unsigned short* __restrict__ vpT) {
    int i = blockIdx.x * 256 + threadIdx.x;
    if (i >= BB * SS * DH) return;
    int dh = i & (DH - 1);
    int s  = (i >> 6) & (SS - 1);
    int b  = i >> 17;
    vpT[((size_t)b * DH + dh) * SS + s] = vp[i];
}

// ---------------------------------------------------------------------------
// Linear layer GEMM: C[M,N] = outScale * (A[M,K] @ W[N,K]^T), bf16 in, f32 acc.
// One wave computes a 32x64 C tile: 2 A-frags + 4 shared B-frags, 8 WMMA per
// K=32 step (halves weight-fragment traffic per FLOP vs a 16x64 tile).
// ---------------------------------------------------------------------------
template <bool OUT_BF16>
__global__ __launch_bounds__(256) void k_gemm_lin(const unsigned short* __restrict__ A,
                                                  const unsigned short* __restrict__ W,
                                                  void* __restrict__ Cv,
                                                  int M, int N, int K, int nStrips,
                                                  float outScale) {
    int wave = (int)((blockIdx.x * 256 + threadIdx.x) >> 5);
    int lane = threadIdx.x & 31;
    int mt   = wave / nStrips;
    int ns   = wave % nStrips;
    if (mt * 32 >= M) return;
    int m0 = mt * 32;
    int n0 = ns * 64;

    v8f acc0[4] = {};
    v8f acc1[4] = {};
    for (int k0 = 0; k0 < K; k0 += 32) {
        v16bf a0 = load_a_frag(A, K, m0,      k0, lane);
        v16bf a1 = load_a_frag(A, K, m0 + 16, k0, lane);
#pragma unroll
        for (int t = 0; t < 4; ++t) {
            v16bf b = load_b_frag(W, K, n0 + t * 16, k0, lane);
            acc0[t] = wmma_bf16(a0, b, acc0[t]);
            acc1[t] = wmma_bf16(a1, b, acc1[t]);
        }
    }

    // f32 C layout: lane<16 -> rows 0..7 (VGPR idx), lane>=16 -> rows 8..15.
    int nlo = lane & 15, half = lane >> 4;
#pragma unroll
    for (int t = 0; t < 4; ++t) {
#pragma unroll
        for (int r = 0; r < 8; ++r) {
            int n = n0 + t * 16 + nlo;
            int ma = m0 + r + 8 * half;
            int mb = ma + 16;
            float va = acc0[t][r] * outScale;
            float vb = acc1[t][r] * outScale;
            if (OUT_BF16) {
                ((unsigned short*)Cv)[(size_t)ma * N + n] = f2bf(va);
                ((unsigned short*)Cv)[(size_t)mb * N + n] = f2bf(vb);
            } else {
                ((float*)Cv)[(size_t)ma * N + n] = va;
                ((float*)Cv)[(size_t)mb * N + n] = vb;
            }
        }
    }
}

// ---------------------------------------------------------------------------
// Flash attention (MQA: one shared K/V head).
// wave -> (b, h, 16-query-row tile). Online softmax over 64-key blocks:
//   scores: 8 WMMA (four 16-key column tiles x two K=32 chunks of DH=64)
//   PV:     8 WMMA (P re-laid C->A through wave-private 16x64 LDS tile)
// Softmax scale is pre-folded into qp, so the hot loop has no score scaling.
// ---------------------------------------------------------------------------
__global__ __launch_bounds__(256) void k_mqa_flash(const unsigned short* __restrict__ qp,
                                                   const unsigned short* __restrict__ kp,
                                                   const unsigned short* __restrict__ vpT,
                                                   unsigned short* __restrict__ aout) {
    __shared__ unsigned short plds[8][16 * 64];   // 16KB: P staging per wave

    int wib  = threadIdx.x >> 5;
    int lane = threadIdx.x & 31;
    int wave = blockIdx.x * 8 + wib;

    int qt = wave & (SS / 16 - 1);     // 0..127
    int h  = (wave >> 7) & (NH - 1);   // 0..15
    int b  = wave >> 11;               // 0..1

    const unsigned short* qb = qp  + (size_t)b * SS * DD;
    const unsigned short* kb = kp  + (size_t)b * SS * DH;
    const unsigned short* vb = vpT + (size_t)b * DH * SS;

    int half = lane >> 4, nlo = lane & 15;

    // Q tile 16xDH as two A fragments (K chunks 0..31, 32..63); reused all loop.
    v16bf aq0 = load_a_frag(qb, DD, qt * 16, h * DH + 0,  lane);
    v16bf aq1 = load_a_frag(qb, DD, qt * 16, h * DH + 32, lane);

    float mrun[8], lrun[8];
#pragma unroll
    for (int i = 0; i < 8; ++i) { mrun[i] = -1e30f; lrun[i] = 0.0f; }
    v8f o[4] = {};

    unsigned short* myP = &plds[wib][0];

    for (int kpos = 0; kpos < SS; kpos += 64) {
        // ---- scores S = Q @ K^T for 64 keys (4 column tiles) --------------
        v8f s[4];
#pragma unroll
        for (int t = 0; t < 4; ++t) {
            v8f st = {};
            v16bf b0 = load_b_frag(kb, DH, kpos + t * 16, 0,  lane);
            st = wmma_bf16(aq0, b0, st);
            v16bf b1 = load_b_frag(kb, DH, kpos + t * 16, 32, lane);
            st = wmma_bf16(aq1, b1, st);
            s[t] = st;
        }

        // ---- online softmax (rows live across the 16 lanes of a half) -----
        float corr[8];
#pragma unroll
        for (int i = 0; i < 8; ++i) {
            float v0 = s[0][i], v1 = s[1][i], v2 = s[2][i], v3 = s[3][i];
            float mx = fmaxf(fmaxf(v0, v1), fmaxf(v2, v3));
#pragma unroll
            for (int d = 1; d < 16; d <<= 1) mx = fmaxf(mx, __shfl_xor(mx, d, 32));
            float mnew = fmaxf(mrun[i], mx);
            float c    = __expf(mrun[i] - mnew);
            float p0   = __expf(v0 - mnew);
            float p1   = __expf(v1 - mnew);
            float p2   = __expf(v2 - mnew);
            float p3   = __expf(v3 - mnew);
            float rs   = (p0 + p1) + (p2 + p3);
#pragma unroll
            for (int d = 1; d < 16; d <<= 1) rs += __shfl_xor(rs, d, 32);
            lrun[i] = lrun[i] * c + rs;
            mrun[i] = mnew;
            corr[i] = c;
            // stage P (bf16, row-major 16x64) for the C->A relayout
            int m = i + 8 * half;
            myP[m * 64 + nlo]      = f2bf(p0);
            myP[m * 64 + 16 + nlo] = f2bf(p1);
            myP[m * 64 + 32 + nlo] = f2bf(p2);
            myP[m * 64 + 48 + nlo] = f2bf(p3);
        }

        // rescale running output by per-row correction (layouts align)
#pragma unroll
        for (int t = 0; t < 4; ++t)
#pragma unroll
            for (int i = 0; i < 8; ++i) o[t][i] *= corr[i];

        // ---- PV: O += P @ V (two K=32 chunks of the 64-key block) ---------
        v16bf pa0 = load_a_frag(myP, 64, 0, 0,  lane);   // wave-synchronous LDS
        v16bf pa1 = load_a_frag(myP, 64, 0, 32, lane);
#pragma unroll
        for (int t = 0; t < 4; ++t) {
            v16bf bv0 = load_b_frag(vb, SS, t * 16, kpos,      lane);
            o[t] = wmma_bf16(pa0, bv0, o[t]);
            v16bf bv1 = load_b_frag(vb, SS, t * 16, kpos + 32, lane);
            o[t] = wmma_bf16(pa1, bv1, o[t]);
        }
    }

    // ---- epilogue: divide by l, write [B,S,H*DH] bf16 ---------------------
    unsigned short* ob = aout + (size_t)b * SS * DD;
#pragma unroll
    for (int i = 0; i < 8; ++i) {
        float inv = 1.0f / lrun[i];
        int m = qt * 16 + i + 8 * half;
#pragma unroll
        for (int t = 0; t < 4; ++t)
            ob[(size_t)m * DD + h * DH + t * 16 + nlo] = f2bf(o[t][i] * inv);
    }
}

// ---------------------------------------------------------------------------
// launch
// ---------------------------------------------------------------------------
extern "C" void kernel_launch(void* const* d_in, const int* in_sizes, int n_in,
                              void* d_out, int out_size, void* d_ws, size_t ws_size,
                              hipStream_t stream) {
    (void)in_sizes; (void)n_in; (void)out_size; (void)ws_size;

    const float* q    = (const float*)d_in[0];
    const float* k    = (const float*)d_in[1];
    const float* v    = (const float*)d_in[2];
    const float* w_q  = (const float*)d_in[3];
    const float* w_k  = (const float*)d_in[4];
    const float* w_v  = (const float*)d_in[5];
    const float* w_o  = (const float*)d_in[6];

    // ---- workspace carve (bf16 halves), ~46 MiB total ----
    unsigned short* p = (unsigned short*)d_ws;
    unsigned short* qbf   = p; p += (size_t)MM * DD;       // 8 MiB
    unsigned short* kbf   = p; p += (size_t)MM * DD;       // 8 MiB
    unsigned short* vbf   = p; p += (size_t)MM * DD;       // 8 MiB
    unsigned short* wqbf  = p; p += (size_t)DD * DD;       // 2 MiB
    unsigned short* wkbf  = p; p += (size_t)DH * DD;       // 128 KiB
    unsigned short* wvbf  = p; p += (size_t)DH * DD;       // 128 KiB
    unsigned short* wobf  = p; p += (size_t)DD * DD;       // 2 MiB
    unsigned short* qpbf  = p; p += (size_t)MM * DD;       // 8 MiB
    unsigned short* kpbf  = p; p += (size_t)MM * DH;       // 512 KiB
    unsigned short* vpbf  = p; p += (size_t)MM * DH;       // 512 KiB
    unsigned short* vptbf = p; p += (size_t)MM * DH;       // 512 KiB
    unsigned short* atbf  = p; p += (size_t)MM * DD;       // 8 MiB

    auto cvt = [&](const float* src, unsigned short* dst, int n) {
        int n4 = n / 4;
        k_f32_to_bf16x4<<<(n4 + 255) / 256, 256, 0, stream>>>(
            (const float4*)src, (uint2*)dst, n4);
    };
    cvt(q,   qbf,  MM * DD);
    cvt(k,   kbf,  MM * DD);
    cvt(v,   vbf,  MM * DD);
    cvt(w_q, wqbf, DD * DD);
    cvt(w_k, wkbf, DH * DD);
    cvt(w_v, wvbf, DH * DD);
    cvt(w_o, wobf, DD * DD);

    // projections. qp carries the softmax scale 1/sqrt(DH)=2^-3 (exact in bf16).
    // waves = (M/32) * (N/64); 8 waves per 256-thread block.
    k_gemm_lin<true><<<256, 256, 0, stream>>>(qbf, wqbf, qpbf, MM, DD, DD, DD / 64, 0.125f);
    k_gemm_lin<true><<<16,  256, 0, stream>>>(kbf, wkbf, kpbf, MM, DH, DD, 1, 1.0f);
    k_gemm_lin<true><<<16,  256, 0, stream>>>(vbf, wvbf, vpbf, MM, DH, DD, 1, 1.0f);

    k_transpose_vp<<<(BB * SS * DH + 255) / 256, 256, 0, stream>>>(vpbf, vptbf);

    // flash attention: 4096 waves = (B=2)*(H=16)*(S/16=128)
    k_mqa_flash<<<512, 256, 0, stream>>>(qpbf, kpbf, vptbf, atbf);

    // output projection -> f32 d_out
    k_gemm_lin<false><<<256, 256, 0, stream>>>(atbf, wobf, d_out, MM, DD, DD, DD / 64, 1.0f);
}
